// Ours_91233695302042
// MI455X (gfx1250) — matrix operand order repaired and checked by reading, not compiled, not run on hardware
//
#include <hip/hip_runtime.h>
#include <hip/hip_bf16.h>
#include <cstdint>

typedef __bf16 bf16_t;
typedef __attribute__((ext_vector_type(16))) __bf16 v16bf;
typedef __attribute__((ext_vector_type(8)))  __bf16 v8bf;
typedef __attribute__((ext_vector_type(8)))  float  v8f;

#define BATCH 256
#define INP   2048
#define C1    1024
#define C2    512
#define HW    49
#define NPAD  64
#define TOPN  4
#define NANCH 294

// ---- WMMA fragment loaders (bf16 16x16x32, wave32) ------------------------
// A operand (16x32, M x K): lane l holds row (l&15); element j of the v16bf
// maps to K = k0 + (j<8?0:16) + (l>>4)*8 + (j&7)  [ISA 7.12.2]
__device__ __forceinline__ v16bf load_a_frag(const bf16_t* rowp, int k0, int hs) {
  const bf16_t* p = rowp + k0 + hs * 8;
  v8bf lo = *(const v8bf*)(p);
  v8bf hi = *(const v8bf*)(p + 16);
  v16bf r;
#pragma unroll
  for (int j = 0; j < 8; ++j) { r[j] = lo[j]; r[j + 8] = hi[j]; }
  return r;
}
// B operand (32x16, K x N): lane l holds column (l&15); element j maps to
// K = k0 + (j<8?0:16) + (l>>4)*8 + (j&7). colp points at a [N][K] row (K contiguous).
__device__ __forceinline__ v16bf load_b_frag(const bf16_t* colp, int k0, int hs) {
  const bf16_t* p = colp + k0 + hs * 8;
  v8bf lo = *(const v8bf*)(p);
  v8bf hi = *(const v8bf*)(p + 16);
  v16bf r;
#pragma unroll
  for (int j = 0; j < 8; ++j) { r[j] = lo[j]; r[j + 8] = hi[j]; }
  return r;
}

// ---- prep: bf16 weight conversion + BN folding ----------------------------
__global__ void prep_kernel(const float* __restrict__ w_tri, const float* __restrict__ b_tri,
                            const float* __restrict__ gamma, const float* __restrict__ beta,
                            const float* __restrict__ rmean, const float* __restrict__ rvar,
                            const float* __restrict__ w_down,
                            bf16_t* __restrict__ w1, bf16_t* __restrict__ w2,
                            float* __restrict__ scale, float* __restrict__ shift) {
  size_t i = (size_t)blockIdx.x * blockDim.x + threadIdx.x;
  size_t stride = (size_t)gridDim.x * blockDim.x;
  const size_t N1 = (size_t)C1 * INP;       // w_tri elements
  const size_t N2 = (size_t)9 * C2 * C1;    // w_down elements (reordered)
  for (size_t j = i; j < N2; j += stride) {
    if (j < N1) w1[j] = (bf16_t)w_tri[j];
    // w2[p][o][c] = w_down[o][c][ky][kx] with p = ky*3+kx
    size_t p = j / ((size_t)C2 * C1);
    size_t rem = j % ((size_t)C2 * C1);
    size_t o = rem / C1, c = rem % C1;
    w2[j] = (bf16_t)w_down[(o * C1 + c) * 9 + p];
    if (j < C1) {
      float rs = rsqrtf(rvar[j] + 1e-5f);
      float sc = gamma[j] * rs;
      scale[j] = sc;
      shift[j] = (b_tri[j] - rmean[j]) * sc + beta[j];
    }
  }
}

// ---- x [B,2048,49] fp32 -> xT [B,64,2048] bf16 (N padded, zero-filled) ----
// Stages through LDS using CDNA5 async global->LDS copies (ASYNCcnt).
__global__ void transpose_x_kernel(const float* __restrict__ x, bf16_t* __restrict__ xT) {
  __shared__ float tile[64][51];   // 51 stride: conflict-free transposed reads
  int b = blockIdx.x;
  int c0 = blockIdx.y * 64;
  int t = threadIdx.x;
  for (int e = t; e < 64 * HW; e += 256) {
    int ci = e / HW, hw = e % HW;
    unsigned long long gp =
        (unsigned long long)(uintptr_t)&x[((size_t)b * INP + c0 + ci) * HW + hw];
    unsigned lds_off = (unsigned)(uintptr_t)&tile[ci][hw];  // LDS aperture: addr[31:0]
    asm volatile("global_load_async_to_lds_b32 %0, %1, off"
                 :: "v"(lds_off), "v"(gp) : "memory");
  }
  asm volatile("s_wait_asynccnt 0x0" ::: "memory");
  __syncthreads();
  for (int e = t; e < 64 * 64; e += 256) {
    int hw = e / 64, ci = e % 64;
    float v = (hw < HW) ? tile[ci][hw] : 0.0f;
    xT[((size_t)b * NPAD + hw) * INP + c0 + ci] = (bf16_t)v;
  }
}

// ---- GEMM1: h = relu(scale*(w_tri @ x) + shift); writes hF(fp32), hT(bf16 T)
// grid (256, 4), block 256 (8 waves). Wave -> one 64x32 output tile
// (4 M-tiles x 2 N-tiles), sharing each A fragment across both N-tiles.
__global__ void gemm1_kernel(const bf16_t* __restrict__ w1, const bf16_t* __restrict__ xT,
                             const float* __restrict__ scale, const float* __restrict__ shift,
                             float* __restrict__ hF, bf16_t* __restrict__ hT) {
  int b = blockIdx.x;
  int tid = threadIdx.x;
  int wid = tid >> 5, lane = tid & 31, l15 = lane & 15, hs = lane >> 4;
  int W = blockIdx.y * 8 + wid;     // 0..31
  int mblock = W >> 1;              // 0..15 -> 64 rows each
  int npair  = W & 1;               // 0..1 -> n-tiles {2p, 2p+1}
  int Mbase = mblock * 64;
  int n0 = (npair * 2) * 16 + l15;  // padded spatial columns
  int n1 = n0 + 16;
  const bf16_t* bcol0 = xT + ((size_t)b * NPAD + n0) * INP;
  const bf16_t* bcol1 = xT + ((size_t)b * NPAD + n1) * INP;
  v8f acc[2][4] = {};
  for (int k0 = 0; k0 < INP; k0 += 32) {
    v16bf bf0 = load_b_frag(bcol0, k0, hs);
    v16bf bf1 = load_b_frag(bcol1, k0, hs);
#pragma unroll
    for (int tI = 0; tI < 4; ++tI) {
      const bf16_t* arow = w1 + (size_t)(Mbase + tI * 16 + l15) * INP;
      v16bf af = load_a_frag(arow, k0, hs);
      acc[0][tI] = __builtin_amdgcn_wmma_f32_16x16x32_bf16(
          false, af, false, bf0, (short)0, acc[0][tI], false, false);
      acc[1][tI] = __builtin_amdgcn_wmma_f32_16x16x32_bf16(
          false, af, false, bf1, (short)0, acc[1][tI], false, false);
    }
  }
  float* hfb = hF + (size_t)b * C1 * HW;
#pragma unroll
  for (int nt = 0; nt < 2; ++nt) {
    int n = nt ? n1 : n0;
    if (n < HW) {
#pragma unroll
      for (int tI = 0; tI < 4; ++tI) {
        int obase = Mbase + tI * 16 + hs * 8;  // 8 consecutive channels per lane
        v8bf hrow;
#pragma unroll
        for (int r = 0; r < 8; ++r) {
          int o = obase + r;
          float v = acc[nt][tI][r] * scale[o] + shift[o];
          v = v > 0.0f ? v : 0.0f;
          hfb[(size_t)o * HW + n] = v;
          hrow[r] = (bf16_t)v;
        }
        *(v8bf*)(hT + ((size_t)b * NPAD + n) * C1 + obase) = hrow;  // 16B store
      }
    }
  }
}

// ---- GEMM2: 3x3 conv as 9 shifted 1x1 GEMMs over hT; d1 = relu(. + b_down)
// grid (256, 2), block 256 (8 waves). Wave -> one 64x32 output tile.
__global__ void gemm2_kernel(const bf16_t* __restrict__ w2, const bf16_t* __restrict__ hT,
                             const float* __restrict__ b_down, float* __restrict__ d1) {
  int b = blockIdx.x;
  int tid = threadIdx.x;
  int wid = tid >> 5, lane = tid & 31, l15 = lane & 15, hs = lane >> 4;
  int W = blockIdx.y * 8 + wid;     // 0..15
  int mblock = W >> 1;              // 0..7
  int npair  = W & 1;
  int Mbase = mblock * 64;
  int n0 = (npair * 2) * 16 + l15;
  int n1 = n0 + 16;
  int y0c = n0 / 7, x0c = n0 % 7;
  int y1c = n1 / 7, x1c = n1 % 7;
  const bf16_t* hTb = hT + (size_t)b * NPAD * C1;
  v16bf zero16;
#pragma unroll
  for (int j = 0; j < 16; ++j) zero16[j] = (bf16_t)0.0f;
  v8f acc[2][4] = {};
  for (int p = 0; p < 9; ++p) {
    int dy = p / 3 - 1, dx = p % 3 - 1;
    int sy0 = y0c + dy, sx0 = x0c + dx;
    int sy1 = y1c + dy, sx1 = x1c + dx;
    bool valid0 = (n0 < HW) && sy0 >= 0 && sy0 < 7 && sx0 >= 0 && sx0 < 7;
    bool valid1 = (n1 < HW) && sy1 >= 0 && sy1 < 7 && sx1 >= 0 && sx1 < 7;
    const bf16_t* bcol0 = hTb + (size_t)(valid0 ? (sy0 * 7 + sx0) : 0) * C1;
    const bf16_t* bcol1 = hTb + (size_t)(valid1 ? (sy1 * 7 + sx1) : 0) * C1;
    const bf16_t* wA = w2 + (size_t)p * C2 * C1;
    for (int k0 = 0; k0 < C1; k0 += 32) {
      v16bf bf0 = valid0 ? load_b_frag(bcol0, k0, hs) : zero16;
      v16bf bf1 = valid1 ? load_b_frag(bcol1, k0, hs) : zero16;
#pragma unroll
      for (int tI = 0; tI < 4; ++tI) {
        const bf16_t* arow = wA + (size_t)(Mbase + tI * 16 + l15) * C1;
        v16bf af = load_a_frag(arow, k0, hs);
        acc[0][tI] = __builtin_amdgcn_wmma_f32_16x16x32_bf16(
            false, af, false, bf0, (short)0, acc[0][tI], false, false);
        acc[1][tI] = __builtin_amdgcn_wmma_f32_16x16x32_bf16(
            false, af, false, bf1, (short)0, acc[1][tI], false, false);
      }
    }
  }
  float* d1b = d1 + (size_t)b * C2 * HW;
#pragma unroll
  for (int nt = 0; nt < 2; ++nt) {
    int n = nt ? n1 : n0;
    if (n < HW) {
#pragma unroll
      for (int tI = 0; tI < 4; ++tI) {
        int obase = Mbase + tI * 16 + hs * 8;
#pragma unroll
        for (int r = 0; r < 8; ++r) {
          int o = obase + r;
          float v = acc[nt][tI][r] + b_down[o];
          d1b[(size_t)o * HW + n] = v > 0.0f ? v : 0.0f;
        }
      }
    }
  }
}

// ---- scores: [B,294] = w_tidy(6x512) @ d1 + b_tidy ------------------------
__global__ void scores_kernel(const float* __restrict__ d1, const float* __restrict__ w_tidy,
                              const float* __restrict__ b_tidy, float* __restrict__ scores) {
  int b = blockIdx.x, t = threadIdx.x;
  if (t >= NANCH) return;
  int j = t / HW, n = t % HW;
  const float* db = d1 + (size_t)b * C2 * HW + n;
  const float* wj = w_tidy + j * C2;
  float s = b_tidy[j];
  for (int c = 0; c < C2; ++c) s += wj[c] * db[(size_t)c * HW];
  scores[b * NANCH + t] = s;
}

// ---- greedy NMS, top-4, IoU >= 0.25 suppress; lowest-index tie-break ------
__global__ void nms_kernel(const float* __restrict__ scores, const int* __restrict__ anchors,
                           int* __restrict__ sel) {
  __shared__ float sc[NANCH];
  __shared__ float area[NANCH];
  __shared__ unsigned char alive[NANCH];
  __shared__ float rv[128];
  __shared__ int   ri[128];
  int b = blockIdx.x, t = threadIdx.x;
  for (int i = t; i < NANCH; i += 128) {
    sc[i] = scores[b * NANCH + i];
    float y0 = (float)anchors[4*i], x0 = (float)anchors[4*i+1];
    float y1 = (float)anchors[4*i+2], x1 = (float)anchors[4*i+3];
    area[i] = (y1 - y0) * (x1 - x0);
    alive[i] = 1;
  }
  __syncthreads();
  for (int round = 0; round < TOPN; ++round) {
    float bv = -__builtin_huge_valf(); int bi = NANCH;
    for (int i = t; i < NANCH; i += 128) {
      float v = alive[i] ? sc[i] : -__builtin_huge_valf();
      if (v > bv || (v == bv && i < bi)) { bv = v; bi = i; }
    }
    rv[t] = bv; ri[t] = bi;
    __syncthreads();
    for (int s = 64; s > 0; s >>= 1) {
      if (t < s) {
        if (rv[t+s] > rv[t] || (rv[t+s] == rv[t] && ri[t+s] < ri[t])) {
          rv[t] = rv[t+s]; ri[t] = ri[t+s];
        }
      }
      __syncthreads();
    }
    int ci = ri[0];
    if (ci >= NANCH) ci = 0;
    float cy0 = (float)anchors[4*ci],   cx0 = (float)anchors[4*ci+1];
    float cy1 = (float)anchors[4*ci+2], cx1 = (float)anchors[4*ci+3];
    float ca = area[ci];
    for (int i = t; i < NANCH; i += 128) {
      float y0 = (float)anchors[4*i], x0 = (float)anchors[4*i+1];
      float y1 = (float)anchors[4*i+2], x1 = (float)anchors[4*i+3];
      float ly = fminf(y1, cy1) - fmaxf(y0, cy0);
      float lx = fminf(x1, cx1) - fmaxf(x0, cx0);
      float inter = (ly < 0.0f || lx < 0.0f) ? 0.0f : ly * lx;
      float iou = inter / (area[i] + ca - inter);
      if (iou >= 0.25f) alive[i] = 0;
    }
    if (t == 0) {
      alive[ci] = 0;
      sel[(b * TOPN + round) * 4 + 0] = anchors[4*ci];
      sel[(b * TOPN + round) * 4 + 1] = anchors[4*ci+1];
      sel[(b * TOPN + round) * 4 + 2] = anchors[4*ci+2];
      sel[(b * TOPN + round) * 4 + 3] = anchors[4*ci+3];
    }
    __syncthreads();
  }
}

// ---- masked mean pool over replicate-padded h (pad via index clamp) -------
__global__ void pool_kernel(const float* __restrict__ hF, const int* __restrict__ sel,
                            float* __restrict__ out) {
  int bt = blockIdx.x;                      // 0..B*TOPN-1
  int c = blockIdx.y * 256 + threadIdx.x;   // channel
  int b = bt >> 2;
  int y0 = sel[bt*4+0], x0 = sel[bt*4+1], y1 = sel[bt*4+2], x1 = sel[bt*4+3];
  int ys = y0 > 0 ? y0 : 0, ye = y1 < 9 ? y1 : 9;
  int xs = x0 > 0 ? x0 : 0, xe = x1 < 9 ? x1 : 9;
  const float* hb = hF + ((size_t)b * C1 + c) * HW;
  float s = 0.0f; int cnt = 0;
  for (int y = ys; y < ye; ++y) {
    int yy = y - 1; yy = yy < 0 ? 0 : (yy > 6 ? 6 : yy);
    for (int x = xs; x < xe; ++x) {
      int xx = x - 1; xx = xx < 0 ? 0 : (xx > 6 ? 6 : xx);
      s += hb[yy * 7 + xx];
      ++cnt;
    }
  }
  out[(size_t)bt * C1 + c] = s / (float)cnt;
}

extern "C" void kernel_launch(void* const* d_in, const int* in_sizes, int n_in,
                              void* d_out, int out_size, void* d_ws, size_t ws_size,
                              hipStream_t stream) {
  const float* x       = (const float*)d_in[0];
  const float* w_tri   = (const float*)d_in[1];
  const float* b_tri   = (const float*)d_in[2];
  const float* gamma   = (const float*)d_in[3];
  const float* beta    = (const float*)d_in[4];
  const float* rmean   = (const float*)d_in[5];
  const float* rvar    = (const float*)d_in[6];
  const float* w_down  = (const float*)d_in[7];
  const float* b_down  = (const float*)d_in[8];
  const float* w_tidy  = (const float*)d_in[9];
  const float* b_tidy  = (const float*)d_in[10];
  const int*   anchors = (const int*)d_in[11];
  float* out = (float*)d_out;

  char* ws = (char*)d_ws;
  size_t off = 0;
  auto take = [&](size_t bytes) {
    char* p = ws + off;
    off = (off + bytes + 255) & ~(size_t)255;
    return p;
  };
  float*  scale = (float*)take((size_t)C1 * 4);
  float*  shift = (float*)take((size_t)C1 * 4);
  bf16_t* w1    = (bf16_t*)take((size_t)C1 * INP * 2);                 //   4 MB
  bf16_t* w2    = (bf16_t*)take((size_t)9 * C2 * C1 * 2);              // 9.4 MB
  bf16_t* xT    = (bf16_t*)take((size_t)BATCH * NPAD * INP * 2);       //  64 MB
  bf16_t* hT    = (bf16_t*)take((size_t)BATCH * NPAD * C1 * 2);        //  32 MB
  float*  hF    = (float*)take((size_t)BATCH * C1 * HW * 4);           //  51 MB
  float*  d1    = (float*)take((size_t)BATCH * C2 * HW * 4);           //  26 MB
  float*  scB   = (float*)take((size_t)BATCH * NANCH * 4);
  int*    selB  = (int*)take((size_t)BATCH * TOPN * 4 * 4);

  prep_kernel<<<dim3(4096), dim3(256), 0, stream>>>(
      w_tri, b_tri, gamma, beta, rmean, rvar, w_down, w1, w2, scale, shift);
  transpose_x_kernel<<<dim3(BATCH, INP / 64), dim3(256), 0, stream>>>(x, xT);
  gemm1_kernel<<<dim3(BATCH, 4), dim3(256), 0, stream>>>(w1, xT, scale, shift, hF, hT);
  gemm2_kernel<<<dim3(BATCH, 2), dim3(256), 0, stream>>>(w2, hT, b_down, d1);
  scores_kernel<<<dim3(BATCH), dim3(320), 0, stream>>>(d1, w_tidy, b_tidy, scB);
  nms_kernel<<<dim3(BATCH), dim3(128), 0, stream>>>(scB, anchors, selB);
  pool_kernel<<<dim3(BATCH * TOPN, C1 / 256), dim3(256), 0, stream>>>(hF, selB, out);
}
